// MMDLoss_55808805044761
// MI455X (gfx1250) — matrix operand order repaired and checked by reading, not compiled, not run on hardware
//
#include <hip/hip_runtime.h>

#define NGROUP 128
#define NPTS   256
#define DDIM   256
#define CBLK   64            // column-block staged in LDS (4 tile-cols)
#define LSTR   264           // LDS row stride in bf16 elements (256 + 8 pad)
#define NJB    (NPTS / CBLK) // 4
#define SMEM_BYTES (2 * CBLK * LSTR * 2 + 3 * 256 * 4)  // Yhi+Ylo + na+nb+red = 70656

typedef __attribute__((ext_vector_type(16))) __bf16 v16bf;
typedef __attribute__((ext_vector_type(8)))  __bf16 v8bf;
typedef __attribute__((ext_vector_type(8)))  float  v8f;

// Truncation split: hi = top 16 bits of f (exact in fp32), lo = top 16 bits of (f - hi).
// Product error ~2^-16 relative (lo*lo term dropped) -- fp32-class for this reduction.
__device__ __forceinline__ __bf16 bf_hi_trunc(float f, float& rem) {
  unsigned u = __float_as_uint(f);
  rem = f - __uint_as_float(u & 0xFFFF0000u);
  return __builtin_bit_cast(__bf16, (unsigned short)(u >> 16));
}
__device__ __forceinline__ __bf16 bf_trunc(float f) {
  unsigned u = __float_as_uint(f);
  return __builtin_bit_cast(__bf16, (unsigned short)(u >> 16));
}

// One block per (group g, pair p): p=0 -> d(x,y) w=+1 ; p=1 -> d(x,x) w=-0.5 ; p=2 -> d(y,y) w=-0.5
__global__ __launch_bounds__(256) void mmd_energy_kernel(const float* __restrict__ inp,
                                                         const float* __restrict__ tgt,
                                                         float* __restrict__ partial) {
  const int bid = blockIdx.x;
  const int g = bid / 3;
  const int p = bid - 3 * g;
  const size_t goff = (size_t)g * NPTS * DDIM;
  const float* A = (p == 2) ? (tgt + goff) : (inp + goff);
  const float* B = (p == 0) ? (tgt + goff) : A;
  const float wsign = (p == 0) ? 1.0f : -0.5f;

  extern __shared__ __align__(16) char smem_raw[];
  __bf16* Yhi = (__bf16*)smem_raw;                 // [CBLK * LSTR]
  __bf16* Ylo = Yhi + CBLK * LSTR;                 // [CBLK * LSTR]
  float*  na  = (float*)(Ylo + CBLK * LSTR);       // [256]
  float*  nb  = na + NPTS;                         // [256]
  float*  red = nb + NPTS;                         // [256]

  const int tid    = threadIdx.x;
  const int lane   = tid & 31;
  const int wave   = tid >> 5;
  const int laneHi = lane >> 4;     // 0 or 1 (lane half)
  const int lm     = lane & 15;

  // ---- per-row squared norms: thread t handles row t of A and of B ----
  {
    const float4* A4 = (const float4*)A;
    const float4* B4 = (const float4*)B;
    float sa = 0.f, sb = 0.f;
#pragma unroll 4
    for (int k = 0; k < DDIM / 4; ++k) {
      float4 va = A4[tid * (DDIM / 4) + k];
      float4 vb = B4[tid * (DDIM / 4) + k];
      sa += va.x * va.x + va.y * va.y + va.z * va.z + va.w * va.w;
      sb += vb.x * vb.x + vb.y * vb.y + vb.z * vb.z + vb.w * vb.w;
    }
    na[tid] = sa;
    nb[tid] = sb;
  }

  float s = 0.f;

#pragma unroll 1
  for (int jb = 0; jb < NJB; ++jb) {
    __syncthreads();   // norms visible (jb==0) / previous-iter LDS reads done

    // ---- stage B rows [jb*64, jb*64+64) into LDS as bf16 hi/lo, packed b64 stores ----
    const float4* B4 = (const float4*)(B + (size_t)jb * CBLK * DDIM);
    if (jb + 1 < NJB)
      __builtin_prefetch(B + (size_t)(jb + 1) * CBLK * DDIM + tid * 4, 0, 1);
#pragma unroll 4
    for (int it = 0; it < CBLK * (DDIM / 4) / 256; ++it) {   // 16 float4 per thread
      int fi  = it * 256 + tid;
      int row = fi >> 6;              // 64 float4 per row
      int c4  = (fi & 63) << 2;       // starting column (multiple of 4)
      float4 v = B4[fi];
      unsigned u0 = __float_as_uint(v.x), u1 = __float_as_uint(v.y);
      unsigned u2 = __float_as_uint(v.z), u3 = __float_as_uint(v.w);
      unsigned h0 = u0 & 0xFFFF0000u, h1 = u1 & 0xFFFF0000u;
      unsigned h2 = u2 & 0xFFFF0000u, h3 = u3 & 0xFFFF0000u;
      unsigned l0 = __float_as_uint(v.x - __uint_as_float(h0));
      unsigned l1 = __float_as_uint(v.y - __uint_as_float(h1));
      unsigned l2 = __float_as_uint(v.z - __uint_as_float(h2));
      unsigned l3 = __float_as_uint(v.w - __uint_as_float(h3));
      uint2 hw, lw;
      hw.x = h1 | (h0 >> 16);
      hw.y = h3 | (h2 >> 16);
      lw.x = (l1 & 0xFFFF0000u) | (l0 >> 16);
      lw.y = (l3 & 0xFFFF0000u) | (l2 >> 16);
      ((uint2*)Yhi)[row * (LSTR / 4) + (c4 >> 2)] = hw;   // LSTR/4 uint2 per row
      ((uint2*)Ylo)[row * (LSTR / 4) + (c4 >> 2)] = lw;
    }
    __syncthreads();

    // ---- each wave owns tile-rows {2w, 2w+1}; 4 tile-cols of this 64-col block ----
#pragma unroll 1
    for (int trI = 0; trI < 2; ++trI) {
      const int tr = wave * 2 + trI;
      const int myrow = tr * 16 + lm;   // A row this lane holds (M = lane%16)
      const float* Arow = A + (size_t)myrow * DDIM;

      v8f acc[4];
#pragma unroll
      for (int tc = 0; tc < 4; ++tc) acc[tc] = (v8f){0.f,0.f,0.f,0.f,0.f,0.f,0.f,0.f};

#pragma unroll 1
      for (int c8 = 0; c8 < 8; ++c8) {
        // A fragment: lane half 0: K = c8*32 + {0..7,16..23}; half 1: K = c8*32 + {8..15,24..31}
        const int k1 = c8 * 32 + laneHi * 8;
        const float4* Ap = (const float4*)(Arow + k1);
        float4 f0 = Ap[0], f1 = Ap[1], f2 = Ap[4], f3 = Ap[5];
        float fa[16] = {f0.x, f0.y, f0.z, f0.w, f1.x, f1.y, f1.z, f1.w,
                        f2.x, f2.y, f2.z, f2.w, f3.x, f3.y, f3.z, f3.w};
        v16bf ahi, alo;
#pragma unroll
        for (int u = 0; u < 16; ++u) {
          float rem;
          __bf16 h = bf_hi_trunc(fa[u], rem);
          __bf16 l = bf_trunc(rem);
          ahi[u] = h;
          alo[u] = l;
        }

        const int kb = c8 * 32 + laneHi * 8;   // B: K = kb + {0..7,16..23} for column N = lane%16
#pragma unroll
        for (int tc = 0; tc < 4; ++tc) {
          const __bf16* ph = &Yhi[(tc * 16 + lm) * LSTR + kb];
          const __bf16* pl = &Ylo[(tc * 16 + lm) * LSTR + kb];
          v8bf bh0 = *(const v8bf*)ph;
          v8bf bh1 = *(const v8bf*)(ph + 16);
          v8bf bl0 = *(const v8bf*)pl;
          v8bf bl1 = *(const v8bf*)(pl + 16);
          v16bf bhi = __builtin_shufflevector(bh0, bh1, 0,1,2,3,4,5,6,7,8,9,10,11,12,13,14,15);
          v16bf blo = __builtin_shufflevector(bl0, bl1, 0,1,2,3,4,5,6,7,8,9,10,11,12,13,14,15);
          acc[tc] = __builtin_amdgcn_wmma_f32_16x16x32_bf16(false, ahi, false, bhi,
                                                            (short)0, acc[tc], false, false);
          acc[tc] = __builtin_amdgcn_wmma_f32_16x16x32_bf16(false, ahi, false, blo,
                                                            (short)0, acc[tc], false, false);
          acc[tc] = __builtin_amdgcn_wmma_f32_16x16x32_bf16(false, alo, false, bhi,
                                                            (short)0, acc[tc], false, false);
        }
      }

      // epilogue: D element r is (M = tr*16 + r + 8*laneHi, N = jb*64 + tc*16 + lane%16)
#pragma unroll
      for (int tc = 0; tc < 4; ++tc) {
        const float nbj = nb[jb * CBLK + tc * 16 + lm];
#pragma unroll
        for (int r = 0; r < 8; ++r) {
          const int mIdx = tr * 16 + r + laneHi * 8;
          float d2 = na[mIdx] + nbj - 2.0f * acc[tc][r];
          d2 = fmaxf(d2, 0.0f);
          s += __builtin_amdgcn_sqrtf(d2);   // raw v_sqrt_f32 (~1 ULP), TRANS op
        }
      }
    }
  }

  // ---- block reduction, weighted partial to workspace (no atomics: deterministic) ----
  red[tid] = s;
  __syncthreads();
#pragma unroll
  for (int off = 128; off > 0; off >>= 1) {
    if (tid < off) red[tid] += red[tid + off];
    __syncthreads();
  }
  if (tid == 0) partial[bid] = red[0] * wsign;
}

// Deterministic fixed-order final sum over 384 partials, then mean normalization.
__global__ void mmd_finalize_kernel(const float* __restrict__ partial, float* __restrict__ out) {
  float s = 0.f;
  for (int i = 0; i < NGROUP * 3; ++i) s += partial[i];
  out[0] = s * (1.0f / (65536.0f * 128.0f));
}

extern "C" void kernel_launch(void* const* d_in, const int* in_sizes, int n_in,
                              void* d_out, int out_size, void* d_ws, size_t ws_size,
                              hipStream_t stream) {
  const float* inp = (const float*)d_in[0];
  const float* tgt = (const float*)d_in[1];
  float* out = (float*)d_out;
  float* partial = (float*)d_ws;   // 384 floats of scratch

  mmd_energy_kernel<<<dim3(NGROUP * 3), dim3(256), SMEM_BYTES, stream>>>(inp, tgt, partial);
  mmd_finalize_kernel<<<dim3(1), dim3(1), 0, stream>>>(partial, out);
}
